// LSTMSelfAttention_67156108640579
// MI455X (gfx1250) — compile-verified, matrix operand order checked by reading, not run on hardware
//
#include <hip/hip_runtime.h>
#include <hip/hip_bf16.h>
#include <math.h>

// Shapes (fixed by the reference)
#define T_DIM 2048
#define B_DIM 16
#define N_DIM 1024
#define ROWS  (T_DIM * B_DIM)   // 32768 rows of the flattened [T*B, N] matrix

// A-tile staging: 64 rows per block, padded row stride so the 16 lanes of a
// fragment read start on distinct 4-bank windows (1032*2 B = 516 dwords,
// 516 % 64 == 4 banks shift per row).
#define AROWS      64
#define ASTRIDE    1032            // bf16 elements per padded row (1024 + 8)

typedef __attribute__((ext_vector_type(16))) __bf16 v16bf;
typedef __attribute__((ext_vector_type(8)))  __bf16 v8bf;
typedef __attribute__((ext_vector_type(8)))  float  v8f;
typedef __attribute__((ext_vector_type(4)))  float  v4f;

__device__ __forceinline__ float fast_tanhf(float x) {
#if __has_builtin(__builtin_amdgcn_tanhf)
    return __builtin_amdgcn_tanhf(x);   // v_tanh_f32 (gfx1250 transcendental)
#else
    return tanhf(x);
#endif
}

// ---------------------------------------------------------------------------
// Fragment loaders following the CDNA5 16-bit WMMA VGPR layout:
//   lanes 0-15 : K = base+0..7  and base+16..23   (VGPR0-3, VGPR4-7)
//   lanes16-31 : K = base+8..15 and base+24..31
// Caller passes p already offset by (kb + sel*8); the second run is +16 elems.
// ---------------------------------------------------------------------------
__device__ __forceinline__ v16bf load_frag_bf16(const __bf16* p) {
    v8bf lo = *(const v8bf*)(p);
    v8bf hi = *(const v8bf*)(p + 16);
    v16bf r;
#pragma unroll
    for (int i = 0; i < 8; ++i) { r[i] = lo[i]; r[8 + i] = hi[i]; }
    return r;
}

__device__ __forceinline__ v8f wmma_bf16(v16bf a, v16bf b, v8f c) {
    return __builtin_amdgcn_wmma_f32_16x16x32_bf16(
        /*neg_a=*/false, a, /*neg_b=*/false, b,
        /*c_mod=*/(short)0, c, /*reuse_a=*/false, /*reuse_b=*/false);
}

// ---------------------------------------------------------------------------
// Kernel 0: convert W (fp32 row-major [m, n]) to bf16 once (2 MB, L2-resident)
// ---------------------------------------------------------------------------
__global__ __launch_bounds__(256)
void k_cvt_w(const float* __restrict__ w, __bf16* __restrict__ wb, int n) {
    int i = blockIdx.x * 256 + threadIdx.x;
    if (i < n) wb[i] = (__bf16)w[i];
}

// ---------------------------------------------------------------------------
// Kernel 1: fused  score[i] = sum_m v[m] * tanh( X[i,:]·W[m,:] + b[m] + X[i,m] )
// 8 waves/block, 64 rows/block staged once into LDS as bf16.
//   waves 0-3: row-tiles 0-3, columns m in [0, 512)
//   waves 4-7: row-tiles 0-3, columns m in [512, 1024)
// Per wave: 8 column-groups x 32 k-steps x 4 WMMAs = 1024 v_wmma.
// ---------------------------------------------------------------------------
__global__ __launch_bounds__(256)
void k_scores(const float* __restrict__ X, const __bf16* __restrict__ Wb,
              const float* __restrict__ bias, const float* __restrict__ vw,
              float* __restrict__ scores)
{
    __shared__ __bf16 Alds[AROWS * ASTRIDE];   // 129 KB
    __shared__ float  sred[2 * AROWS];

    const int tid  = threadIdx.x;
    const int lane = tid & 31;
    const int wave = tid >> 5;
    const int sel  = lane >> 4;          // 0: lanes 0-15, 1: lanes 16-31
    const int l15  = lane & 15;
    const int row_block = blockIdx.x * AROWS;

    // ---- Stage A: 64 rows x 1024 cols fp32 -> bf16 in LDS (coalesced) ----
    // 65536 elements, 8 per thread-iteration, 256 threads -> 32 iterations.
#pragma unroll 4
    for (int it = 0; it < (AROWS * N_DIM) / (256 * 8); ++it) {
        const int chunk = it * 256 + tid;        // 8-element chunks
        const int row   = chunk >> 7;            // 128 chunks per row
        const int kc    = (chunk & 127) * 8;
        const float* gp = X + (size_t)(row_block + row) * N_DIM + kc;
        v4f f0 = *(const v4f*)(gp);
        v4f f1 = *(const v4f*)(gp + 4);
        v8bf o;
#pragma unroll
        for (int i = 0; i < 4; ++i) { o[i] = (__bf16)f0[i]; o[4 + i] = (__bf16)f1[i]; }
        *(v8bf*)(&Alds[row * ASTRIDE + kc]) = o;
    }
    __syncthreads();

    const int rtile = wave & 3;                  // which 16-row tile
    const int hv    = wave >> 2;                 // which m half
    const int row_base = row_block + rtile * 16;
    const int m0 = hv * (N_DIM / 2);
    const int koff = sel * 8;

    const __bf16* Arow = &Alds[(rtile * 16 + l15) * ASTRIDE];

    float sacc[8];
#pragma unroll
    for (int r = 0; r < 8; ++r) sacc[r] = 0.0f;

    for (int mg = 0; mg < (N_DIM / 2); mg += 64) {
        const int mb = m0 + mg;
        // B columns mb..mb+63: column (mb+j+l15) of B == row (mb+j+l15) of W
        const __bf16* Wr0 = Wb + (size_t)(mb + l15) * N_DIM;
        const __bf16* Wr1 = Wr0 + (size_t)16 * N_DIM;
        const __bf16* Wr2 = Wr0 + (size_t)32 * N_DIM;
        const __bf16* Wr3 = Wr0 + (size_t)48 * N_DIM;

        v8f c0 = {}, c1 = {}, c2 = {}, c3 = {};
        for (int kb = 0; kb < N_DIM; kb += 32) {
            v16bf a = load_frag_bf16(Arow + kb + koff);   // LDS, no conversion
            c0 = wmma_bf16(a, load_frag_bf16(Wr0 + kb + koff), c0);
            c1 = wmma_bf16(a, load_frag_bf16(Wr1 + kb + koff), c1);
            c2 = wmma_bf16(a, load_frag_bf16(Wr2 + kb + koff), c2);
            c3 = wmma_bf16(a, load_frag_bf16(Wr3 + kb + koff), c3);
        }

        // Epilogue: bias + residual + tanh + v-weighted accumulation.
        // C layout: VGPR r -> row (r + sel*8), col = chunk_base + l15.
#pragma unroll
        for (int q = 0; q < 4; ++q) {
            const v8f  c  = (q == 0) ? c0 : (q == 1) ? c1 : (q == 2) ? c2 : c3;
            const int  mc = mb + q * 16 + l15;
            const float bs = bias[mc];
            const float vv = vw[mc];
#pragma unroll
            for (int r = 0; r < 8; ++r) {
                const int row = row_base + r + sel * 8;
                const float xres = X[(size_t)row * N_DIM + mc];   // fp32 residual
                sacc[r] += vv * fast_tanhf(c[r] + bs + xres);
            }
        }
    }

    // Reduce across the 16 lanes of each half (wave32): lanes 0-15 hold rows
    // row_base+r, lanes 16-31 hold rows row_base+r+8.
#pragma unroll
    for (int r = 0; r < 8; ++r) {
        float s = sacc[r];
        s += __shfl_xor(s, 1, 32);
        s += __shfl_xor(s, 2, 32);
        s += __shfl_xor(s, 4, 32);
        s += __shfl_xor(s, 8, 32);
        if (l15 == 0)
            sred[hv * AROWS + rtile * 16 + r + sel * 8] = s;
    }
    __syncthreads();

    // Combine the two m-half partials and write 64 scores per block.
    if (tid < AROWS)
        scores[row_block + tid] = sred[tid] + sred[AROWS + tid];
}

// ---------------------------------------------------------------------------
// Kernel 2: masked softmax over T (axis 0), one block per batch column b.
// ---------------------------------------------------------------------------
__global__ __launch_bounds__(256)
void k_softmax(const float* __restrict__ scores,
               const unsigned char* __restrict__ mask,
               float* __restrict__ attn)
{
    __shared__ float red[256];
    const int b = blockIdx.x;
    const int tid = threadIdx.x;
    float sv[T_DIM / 256];

    float mx = -INFINITY;
#pragma unroll
    for (int i = 0; i < T_DIM / 256; ++i) {
        const int t = tid + i * 256;
        float s = scores[t * B_DIM + b];
        if (mask[t * B_DIM + b]) s = -INFINITY;
        sv[i] = s;
        mx = fmaxf(mx, s);
    }
    red[tid] = mx; __syncthreads();
    for (int off = 128; off > 0; off >>= 1) {
        if (tid < off) red[tid] = fmaxf(red[tid], red[tid + off]);
        __syncthreads();
    }
    mx = red[0]; __syncthreads();

    float sum = 0.0f;
#pragma unroll
    for (int i = 0; i < T_DIM / 256; ++i) { sv[i] = __expf(sv[i] - mx); sum += sv[i]; }
    red[tid] = sum; __syncthreads();
    for (int off = 128; off > 0; off >>= 1) {
        if (tid < off) red[tid] += red[tid + off];
        __syncthreads();
    }
    const float inv = 1.0f / red[0];
#pragma unroll
    for (int i = 0; i < T_DIM / 256; ++i)
        attn[(tid + i * 256) * B_DIM + b] = sv[i] * inv;
}

// ---------------------------------------------------------------------------
// Kernel 3: partial weighted sums over T segments (deterministic, no atomics).
// grid = (N/256, B, 16 segments of 128 timesteps)
// ---------------------------------------------------------------------------
__global__ __launch_bounds__(256)
void k_wsum_part(const float* __restrict__ X, const float* __restrict__ attn,
                 float* __restrict__ part)
{
    __shared__ float av[128];
    const int b = blockIdx.y, seg = blockIdx.z;
    const int n = blockIdx.x * 256 + threadIdx.x;
    const int t0 = seg * 128;
    if (threadIdx.x < 128) av[threadIdx.x] = attn[(t0 + threadIdx.x) * B_DIM + b];
    __syncthreads();
    float acc = 0.0f;
#pragma unroll 4
    for (int i = 0; i < 128; ++i)
        acc += av[i] * X[((size_t)(t0 + i) * B_DIM + b) * N_DIM + n];
    part[((size_t)seg * B_DIM + b) * N_DIM + n] = acc;
}

// Kernel 4: reduce the 16 segment partials into z[b, n].
__global__ __launch_bounds__(256)
void k_wsum_reduce(const float* __restrict__ part, float* __restrict__ z)
{
    const int i = blockIdx.x * 256 + threadIdx.x;   // i = b*N + n, 16384 total
    float acc = 0.0f;
#pragma unroll
    for (int s = 0; s < 16; ++s)
        acc += part[(size_t)s * B_DIM * N_DIM + i];
    z[i] = acc;
}

// ---------------------------------------------------------------------------
extern "C" void kernel_launch(void* const* d_in, const int* in_sizes, int n_in,
                              void* d_out, int out_size, void* d_ws, size_t ws_size,
                              hipStream_t stream) {
    const float*         X    = (const float*)d_in[0];          // [T,B,N]
    const unsigned char* mask = (const unsigned char*)d_in[1];  // [T,B] bool
    const float*         W_w  = (const float*)d_in[2];          // [N,N]
    const float*         W_b  = (const float*)d_in[3];          // [N]
    const float*         v_w  = (const float*)d_in[4];          // [1,N]

    float* z_out = (float*)d_out;                    // [B,N]  = 16384 floats
    float* attn  = (float*)d_out + B_DIM * N_DIM;    // [T,B]  = 32768 floats

    // Workspace layout: scores (128 KB) | W bf16 (2 MB) | partials (1 MB)
    char*   ws     = (char*)d_ws;
    float*  scores = (float*)ws;
    __bf16* Wb     = (__bf16*)(ws + (size_t)ROWS * sizeof(float));
    float*  part   = (float*)(ws + (size_t)ROWS * sizeof(float)
                                 + (size_t)N_DIM * N_DIM * sizeof(__bf16));

    k_cvt_w     <<<(N_DIM * N_DIM) / 256, 256, 0, stream>>>(W_w, Wb, N_DIM * N_DIM);
    k_scores    <<<ROWS / AROWS, 256, 0, stream>>>(X, Wb, W_b, v_w, scores);
    k_softmax   <<<B_DIM, 256, 0, stream>>>(scores, mask, attn);
    k_wsum_part <<<dim3(N_DIM / 256, B_DIM, T_DIM / 128), 256, 0, stream>>>(X, attn, part);
    k_wsum_reduce<<<(B_DIM * N_DIM) / 256, 256, 0, stream>>>(part, z_out);
}
